// KittyCatConv_61143154426362
// MI455X (gfx1250) — compile-verified, matrix operand order checked by reading, not run on hardware
//
#include <hip/hip_runtime.h>
#include <hip/hip_bf16.h>

// ---------------------------------------------------------------------------
// Problem constants (from the reference): B=8, H=8, L=LK=1024, DK=64,
// filters (1,3,7,9).  scores are rank-1: scores[q,k] = c * Qtop[q] * Ksort[k]
// with c = dot(wq_back, wk_back) / 8.  Dominant cost = 256MB attn write +
// attn@V GEMM per head (1024x64x1024) -> v_wmma_f32_16x16x32_f16.
// ---------------------------------------------------------------------------

#define Bq   8
#define Hh   8
#define Ll   1024
#define DKk  64
#define BHL  65536          // B*H*L
#define ROWS 64             // B*H
#define ROWLEN 4096         // 4*L per (b,h) row after concat/reshape

typedef __attribute__((ext_vector_type(16))) _Float16 v16h;
typedef __attribute__((ext_vector_type(8)))  float    v8f;

union AFrag { v16h v; _Float16 h[16]; unsigned int u[8]; };
union CFrag { v8f  v; float f[8]; };

// ---------------------------------------------------------------------------
// 1) proj: q[b,h,l] = dot(Q[b,h,l,:], w[0:64])   (blockIdx.y: 0->Q, 1->K)
// ---------------------------------------------------------------------------
__global__ __launch_bounds__(256) void proj_kernel(
    const float* __restrict__ Q, const float* __restrict__ K,
    const float* __restrict__ wq, const float* __restrict__ wk,
    float* __restrict__ qo, float* __restrict__ ko) {
  int idx = blockIdx.x * 256 + threadIdx.x;          // 0..BHL-1
  const float* src = blockIdx.y ? K : Q;
  const float* w   = blockIdx.y ? wk : wq;
  float*       out = blockIdx.y ? ko : qo;
  const float4* s4 = reinterpret_cast<const float4*>(src + (size_t)idx * DKk);
  const float4* w4 = reinterpret_cast<const float4*>(w);
  float acc = 0.f;
#pragma unroll
  for (int i = 0; i < 16; ++i) {
    float4 a = s4[i], b = w4[i];
    acc += a.x * b.x + a.y * b.y + a.z * b.z + a.w * b.w;
  }
  out[idx] = acc;
}

// ---------------------------------------------------------------------------
// 2) conv1d (OIH weights, 'same' padding) + bias, one branch per launch
// ---------------------------------------------------------------------------
__global__ __launch_bounds__(256) void conv_kernel(
    const float* __restrict__ x, const float* __restrict__ w,
    const float* __restrict__ bias, float* __restrict__ y, int F) {
  int idx = blockIdx.x * 256 + threadIdx.x;          // over B*H*L
  int l = idx & 1023;
  int o = (idx >> 10) & 7;
  int b = idx >> 13;
  int p = (F - 1) >> 1;
  float acc = bias[o];
  for (int c = 0; c < Hh; ++c) {
    const float* xr = x + (size_t)(b * Hh + c) * Ll;
    const float* wr = w + (size_t)(o * Hh + c) * F;
    for (int t = 0; t < F; ++t) {
      int pos = l + t - p;
      if (pos >= 0 && pos < Ll) acc += wr[t] * xr[pos];
    }
  }
  y[idx] = acc;
}

// ---------------------------------------------------------------------------
// 3) BN stats: per (src, branch, channel) mean + rsqrt(var+eps) over B*L
// ---------------------------------------------------------------------------
__global__ __launch_bounds__(256) void bn_stats_kernel(
    const float* __restrict__ yQ, const float* __restrict__ yK,
    float* __restrict__ stats) {
  int blk = blockIdx.x;                 // 64 = 2 src * 4 branch * 8 ch
  int src = blk >> 5, br = (blk >> 3) & 3, ch = blk & 7;
  const float* y = (src ? yK : yQ) + (size_t)br * BHL + (size_t)ch * Ll;
  __shared__ float s1[256], s2[256];
  float a = 0.f, b = 0.f;
  for (int i = threadIdx.x; i < Bq * Ll; i += 256) {
    int bb = i >> 10, l = i & 1023;
    float v = y[(size_t)bb * (Hh * Ll) + l];
    a += v; b += v * v;
  }
  s1[threadIdx.x] = a; s2[threadIdx.x] = b;
  __syncthreads();
  for (int st = 128; st > 0; st >>= 1) {
    if (threadIdx.x < st) {
      s1[threadIdx.x] += s1[threadIdx.x + st];
      s2[threadIdx.x] += s2[threadIdx.x + st];
    }
    __syncthreads();
  }
  if (threadIdx.x == 0) {
    float mu  = s1[0] * (1.f / 8192.f);
    float var = s2[0] * (1.f / 8192.f) - mu * mu;
    stats[blk * 2]     = mu;
    stats[blk * 2 + 1] = rsqrtf(var + 1e-5f);
  }
}

// ---------------------------------------------------------------------------
// 4) BN normalize + ELU, in place (gridDim.y = branch, gridDim.z = src)
// ---------------------------------------------------------------------------
__global__ __launch_bounds__(256) void bn_apply_kernel(
    float* __restrict__ yQ, float* __restrict__ yK,
    const float* __restrict__ stats,
    const float* __restrict__ gQ, const float* __restrict__ bQ,
    const float* __restrict__ gK, const float* __restrict__ bK) {
  int idx = blockIdx.x * 256 + threadIdx.x;          // 0..BHL-1 within branch
  int br = blockIdx.y, src = blockIdx.z;
  float* y = (src ? yK : yQ) + (size_t)br * BHL;
  const float* g  = src ? gK : gQ;
  const float* be = src ? bK : bQ;
  int ch = (idx >> 10) & 7;
  int sb = src * 32 + br * 8 + ch;
  float mu = stats[sb * 2], rs = stats[sb * 2 + 1];
  float v = (y[idx] - mu) * rs * g[br * 8 + ch] + be[br * 8 + ch];
  y[idx] = v > 0.f ? v : expm1f(v);                  // ELU(alpha=1)
}

// ---------------------------------------------------------------------------
// 5) Q: top-1024 of each 4096-row, sorted descending (bitonic in LDS)
// ---------------------------------------------------------------------------
__global__ __launch_bounds__(1024) void qtop_kernel(
    const float* __restrict__ y, float* __restrict__ out) {
  __shared__ float buf[ROWLEN];
  const float* src = y + (size_t)blockIdx.x * ROWLEN;
  for (int i = threadIdx.x; i < ROWLEN; i += 1024) buf[i] = src[i];
  __syncthreads();
  for (unsigned k = 2; k <= ROWLEN; k <<= 1)
    for (unsigned j = k >> 1; j > 0; j >>= 1) {
      for (unsigned i = threadIdx.x; i < ROWLEN; i += 1024) {
        unsigned ixj = i ^ j;
        if (ixj > i) {
          float a = buf[i], b = buf[ixj];
          bool desc = ((i & k) == 0);
          if (desc ? (a < b) : (a > b)) { buf[i] = b; buf[ixj] = a; }
        }
      }
      __syncthreads();
    }
  if (threadIdx.x < 1024)
    out[(size_t)blockIdx.x * Ll + threadIdx.x] = buf[threadIdx.x];
}

// ---------------------------------------------------------------------------
// 6) K: mean over 4 scale-chunks, full descending sort of 1024
// ---------------------------------------------------------------------------
__global__ __launch_bounds__(1024) void ksort_kernel(
    const float* __restrict__ y, float* __restrict__ out) {
  __shared__ float buf[Ll];
  const float* src = y + (size_t)blockIdx.x * ROWLEN;
  unsigned i = threadIdx.x;
  buf[i] = 0.25f * (src[i] + src[i + 1024] + src[i + 2048] + src[i + 3072]);
  __syncthreads();
  for (unsigned k = 2; k <= Ll; k <<= 1)
    for (unsigned j = k >> 1; j > 0; j >>= 1) {
      unsigned ixj = i ^ j;
      if (ixj > i) {
        float a = buf[i], b = buf[ixj];
        bool desc = ((i & k) == 0);
        if (desc ? (a < b) : (a > b)) { buf[i] = b; buf[ixj] = a; }
      }
      __syncthreads();
    }
  out[(size_t)blockIdx.x * Ll + i] = buf[i];
}

// ---------------------------------------------------------------------------
// 7) Fused attention: rank-1 scores -> softmax -> attn write + WMMA attn@V.
//    grid = (64 heads, 16 q-blocks of 64 rows), 128 threads = 4 waves.
//    Each wave: 16 q-rows x 64 d-cols = 4 f32 16x16 accumulators, K step 32.
//    A frag (probabilities) generated in registers per the 16-bit 16x32 A
//    layout; B frag read as b32 pairs from LDS-transposed V tile.
//    Z is inverted once per row so the 256MB attn stream is exp*mul*store
//    (no per-element IEEE division expansion).
// ---------------------------------------------------------------------------
__global__ __launch_bounds__(128) void attn_kernel(
    const float* __restrict__ Qtop, const float* __restrict__ Ksort,
    const float* __restrict__ V, const float* __restrict__ wqb,
    const float* __restrict__ wkb, float* __restrict__ ctx,
    float* __restrict__ attn) {
  __shared__ float sKs[Ll];
  __shared__ float sSq[64], sMq[64], sZ[64];         // sZ -> 1/Z after inversion
  __shared__ __align__(16) _Float16 sVt[DKk * 32];   // V^T tile [d][k]
  __shared__ float sC;

  const int tid  = threadIdx.x;
  const int head = blockIdx.x;                       // 0..63
  const int qb   = blockIdx.y * 64;

  if (tid == 0) {
    float c = 0.f;
    for (int d = 0; d < DKk; ++d) c += wqb[d] * wkb[d];
    sC = c * 0.125f;                                 // /sqrt(64)
  }
  for (int i = tid; i < Ll; i += 128) sKs[i] = Ksort[(size_t)head * Ll + i];
  if (tid < 64) sZ[tid] = 0.f;
  __syncthreads();

  if (tid < 64) {
    float sq = sC * Qtop[(size_t)head * Ll + qb + tid];
    sSq[tid] = sq;
    // Ksort descending: row max at k=0 if sq>=0 else k=1023
    sMq[tid] = (sq >= 0.f) ? sq * sKs[0] : sq * sKs[Ll - 1];
  }
  __syncthreads();

  { // softmax denominator (2 threads per row, LDS float atomics)
    int r = tid & 63, half = tid >> 6;
    float sq = sSq[r], mq = sMq[r], z = 0.f;
    for (int k = half * 512; k < half * 512 + 512; ++k)
      z += __expf(sq * sKs[k] - mq);
    atomicAdd(&sZ[r], z);
  }
  __syncthreads();
  if (tid < 64) sZ[tid] = 1.f / sZ[tid];             // one divide per row
  __syncthreads();

  { // coalesced attn write: 64 rows x 1024 (bulk of the 256MB traffic)
    size_t base = ((size_t)head << 20) + ((size_t)qb << 10);
    for (int idx = tid; idx < 64 * Ll; idx += 128) {
      int r = idx >> 10, k = idx & 1023;
      attn[base + ((size_t)r << 10) + k] =
          __expf(sSq[r] * sKs[k] - sMq[r]) * sZ[r];
    }
  }

  // ---- WMMA GEMM: context = P @ V -------------------------------------
  const int lane = tid & 31, wave = tid >> 5;
  const int m = lane & 15, hi = lane >> 4;
  const int rloc = wave * 16 + m;
  const float sqm = sSq[rloc], mqm = sMq[rloc];

  CFrag acc[4];
#pragma unroll
  for (int nt = 0; nt < 4; ++nt)
#pragma unroll
    for (int j = 0; j < 8; ++j) acc[nt].f[j] = 0.f;

  const float* Vh = V + ((size_t)head << 16);        // head*1024*64
  for (int kk = 0; kk < Ll; kk += 32) {
    __syncthreads();
    for (int i = tid; i < 32 * DKk; i += 128) {      // stage V^T tile as f16
      int k = i >> 6, d = i & 63;
      sVt[d * 32 + k] = (_Float16)Vh[(size_t)(kk + k) * DKk + d];
    }
    __syncthreads();

    AFrag a;
#pragma unroll
    for (int v = 0; v < 8; ++v) {
      int K0 = ((v < 4) ? 2 * v : 16 + 2 * (v - 4)) + (hi ? 8 : 0);
      a.h[2 * v]     = (_Float16)__expf(sqm * sKs[kk + K0]     - mqm);
      a.h[2 * v + 1] = (_Float16)__expf(sqm * sKs[kk + K0 + 1] - mqm);
    }
#pragma unroll
    for (int nt = 0; nt < 4; ++nt) {
      AFrag b;
#pragma unroll
      for (int v = 0; v < 8; ++v) {
        int K0 = ((v < 4) ? 2 * v : 16 + 2 * (v - 4)) + (hi ? 8 : 0);
        b.u[v] = *reinterpret_cast<const unsigned int*>(
            &sVt[(nt * 16 + (lane & 15)) * 32 + K0]);   // aligned (K,K+1) pair
      }
      acc[nt].v = __builtin_amdgcn_wmma_f32_16x16x32_f16(
          false, a.v, false, b.v, (short)0, acc[nt].v, false, false);
    }
    if (kk + 32 < Ll) {                                // prefetch next V tile
      __builtin_prefetch(&Vh[(size_t)(kk + 32 + lane) * DKk], 0, 1);
      __builtin_prefetch(&Vh[(size_t)(kk + 32 + lane) * DKk + 32], 0, 1);
    }
  }

  // D layout: VGPR i -> row M=i+8*hi, lanes 0..15 -> N
#pragma unroll
  for (int i = 0; i < 8; ++i) {
    int qr = wave * 16 + i + (hi ? 8 : 0);
    float invz = sZ[qr];                             // already 1/Z
#pragma unroll
    for (int nt = 0; nt < 4; ++nt) {
      int d = nt * 16 + (lane & 15);
      ctx[((size_t)head << 16) + (size_t)(qb + qr) * DKk + d] =
          acc[nt].f[i] * invz;
    }
  }
}

// ---------------------------------------------------------------------------
// launch
// ---------------------------------------------------------------------------
extern "C" void kernel_launch(void* const* d_in, const int* in_sizes, int n_in,
                              void* d_out, int out_size, void* d_ws, size_t ws_size,
                              hipStream_t stream) {
  (void)in_sizes; (void)n_in; (void)out_size; (void)ws_size;
  const float* Q   = (const float*)d_in[0];
  const float* K   = (const float*)d_in[1];
  const float* V   = (const float*)d_in[2];
  // d_in[3] = attn_mask (unused, matching reference)
  const float* pqw = (const float*)d_in[4];
  const float* pkw = (const float*)d_in[5];
  const float* wqb = (const float*)d_in[6];   // proj_back_q_w [64]
  const float* wkb = (const float*)d_in[7];   // proj_back_k_w [64]
  const float* cqw[4] = {(const float*)d_in[8],  (const float*)d_in[9],
                         (const float*)d_in[10], (const float*)d_in[11]};
  const float* cqb[4] = {(const float*)d_in[12], (const float*)d_in[13],
                         (const float*)d_in[14], (const float*)d_in[15]};
  const float* ckw[4] = {(const float*)d_in[16], (const float*)d_in[17],
                         (const float*)d_in[18], (const float*)d_in[19]};
  const float* ckb[4] = {(const float*)d_in[20], (const float*)d_in[21],
                         (const float*)d_in[22], (const float*)d_in[23]};
  const float* gQ = (const float*)d_in[24];
  const float* bQ = (const float*)d_in[25];
  const float* gK = (const float*)d_in[26];
  const float* bK = (const float*)d_in[27];

  float* ctx  = (float*)d_out;                       // [8,8,1024,64]
  float* attn = ctx + (size_t)Bq * Hh * Ll * DKk;    // [8,8,1024,1024]

  float* ws    = (float*)d_ws;
  float* qproj = ws;                                 // 65536
  float* kproj = ws + 65536;                         // 65536
  float* yQ    = ws + 131072;                        // 4*65536 (branch-major)
  float* yK    = ws + 393216;                        // 4*65536
  float* stats = ws + 655360;                        // 128
  float* Qtop  = ws + 655488;                        // 65536
  float* Ksrt  = ws + 721024;                        // 65536

  proj_kernel<<<dim3(256, 2), 256, 0, stream>>>(Q, K, pqw, pkw, qproj, kproj);

  const int F[4] = {1, 3, 7, 9};
  for (int br = 0; br < 4; ++br) {
    conv_kernel<<<256, 256, 0, stream>>>(qproj, cqw[br], cqb[br],
                                         yQ + (size_t)br * BHL, F[br]);
    conv_kernel<<<256, 256, 0, stream>>>(kproj, ckw[br], ckb[br],
                                         yK + (size_t)br * BHL, F[br]);
  }
  bn_stats_kernel<<<64, 256, 0, stream>>>(yQ, yK, stats);
  bn_apply_kernel<<<dim3(256, 4, 2), 256, 0, stream>>>(yQ, yK, stats,
                                                       gQ, bQ, gK, bK);
  qtop_kernel<<<ROWS, 1024, 0, stream>>>(yQ, Qtop);
  ksort_kernel<<<ROWS, 1024, 0, stream>>>(yK, Ksrt);
  attn_kernel<<<dim3(ROWS, Ll / 64), 128, 0, stream>>>(Qtop, Ksrt, V, wqb, wkb,
                                                       ctx, attn);
}
